// DistributedDotGAT_47897475285035
// MI455X (gfx1250) — matrix operand order repaired and checked by reading, not compiled, not run on hardware
//
#include <hip/hip_runtime.h>
#include <hip/hip_bf16.h>
#include <math.h>

// ---------------------------------------------------------------------------
// DistributedDotGAT for MI455X (gfx1250): bf16 WMMA (f32 accumulate) GEMMs,
// LDS-staged attention, fused SiLU/bias/residual epilogue.
// Weight-bandwidth bound (~1.35GB fp32 weights/call @ 23.3TB/s ≈ 58us floor).
//
// WMMA operand layouts (ISA 7.12.2/7.12.4, v_wmma_f32_16x16x32_bf16):
//  A (16x32): lane l<16 -> row l, K {0..7,16..23}; lane l+16 -> row l,
//             K {8..15,24..31}; element e<8 -> K kb+e, e>=8 -> K 16+kb+(e-8).
//  B (32x16) as loaded here (A uses the same K order): lane l<16 -> col l, K 0..15 (element e -> K e);
//             lane l+16 -> col l, K 16..31.  (B fragment = 16 contiguous K)
//  C/D (16x16 f32): lane = N (mod 16), VGPR r -> M = r + (lane>=16 ? 8 : 0).
// LDS tiles are stored so every fragment load is contiguous (ds_load_b128).
// ---------------------------------------------------------------------------

#define BATCH 32
#define AGENTS 128
#define DMODEL 512
#define NHEADS 8
#define DHEAD 64
#define ROWSTRIDE_B ((long)AGENTS * DMODEL)   // 65536 floats between batches
#define NROWS (BATCH * AGENTS)                // 4096 LN rows
#define HSIZE ((long)BATCH * AGENTS * DMODEL) // 2097152 floats

typedef __attribute__((ext_vector_type(8)))  float  v8f;
typedef __attribute__((ext_vector_type(16))) __bf16 v16bf;

static __device__ __forceinline__ v8f wmma_bf16(v16bf a, v16bf b, v8f c) {
    return __builtin_amdgcn_wmma_f32_16x16x32_bf16(
        /*neg_a=*/false, a, /*neg_b=*/false, b,
        /*c_mod=*/(short)0, c, /*reuse_a=*/false, /*reuse_b=*/false);
}

// Load one A-operand fragment from a row-major [*][rowStride] bf16 LDS tile.
// rowPtr points at element K=kcBase of the lane's matrix row.
static __device__ __forceinline__ v16bf load_afrag(const __bf16* rowPtr, int kb) {
    v16bf a;
    const __bf16* p = rowPtr + 2 * kb;    // kb = (lane>>4)*8 -> K base (lane>>4)*16
    #pragma unroll
    for (int e = 0; e < 8; ++e) a[e] = p[e];          // K 2kb+0..7
    #pragma unroll
    for (int e = 0; e < 8; ++e) a[8 + e] = p[8 + e];  // K 2kb+8..15
    return a;
}

// Load one B-operand fragment: 16 contiguous K values for this lane.
static __device__ __forceinline__ v16bf load_bfrag(const __bf16* p /* K=kcBase+kb2 */) {
    v16bf b;
    #pragma unroll
    for (int e = 0; e < 16; ++e) b[e] = p[e];
    return b;
}

// ---------------------------------------------------------------------------
// Per-agent GEMM:  C[b,agent,n0+n] (+)= f( sum_k A[b,agent,k] * W[agent,k,n] )
// M = 32 (batch), N tile = 128, K = 512 in chunks of 32.
// flags: 0 = plain store; 1 = +bias, SiLU, accumulate into C (residual).
// wAgentStride = 0 -> shared weight (W_v).
// ---------------------------------------------------------------------------
__global__ __launch_bounds__(256)
void gemm_agent_kernel(const float* __restrict__ Aact,
                       const float* __restrict__ W,
                       const float* __restrict__ bias,
                       float* __restrict__ C,
                       long wAgentStride, int flags)
{
    __shared__ __bf16 sA [32][32];    // 2 KB  (batch-row major: [m][k])
    __shared__ __bf16 sBt[128][32];   // 8 KB  (K-major: [n][k]) -> contiguous B frags

    const int agent = blockIdx.y;
    const int n0    = blockIdx.x * 128;
    const int tid   = threadIdx.x;
    const int lane  = tid & 31;
    const int wave  = tid >> 5;

    const float* aBase = Aact + (long)agent * DMODEL;
    const float* wBase = W + (long)agent * wAgentStride + n0;

    v8f acc0 = {0.f,0.f,0.f,0.f,0.f,0.f,0.f,0.f};
    v8f acc1 = {0.f,0.f,0.f,0.f,0.f,0.f,0.f,0.f};

    const int mrow = lane & 15;
    const int kb   = (lane >> 4) * 8;   // A fragment K base is 2*kb
    const int kb2  = (lane >> 4) * 16;  // B fragment K base
    const int ncol = wave * 16 + mrow;  // this wave's N column (0..127)

    for (int k0 = 0; k0 < DMODEL; k0 += 32) {
        // ---- stage A tile: 32 batch-rows x 32 k (fp32 -> bf16) ----
        {
            int m  = tid >> 3;
            int kk = (tid & 7) * 4;
            float4 f = *(const float4*)(aBase + (long)m * ROWSTRIDE_B + k0 + kk);
            sA[m][kk + 0] = (__bf16)f.x;
            sA[m][kk + 1] = (__bf16)f.y;
            sA[m][kk + 2] = (__bf16)f.z;
            sA[m][kk + 3] = (__bf16)f.w;
        }
        // ---- stage B tile transposed: global [k][n] -> LDS [n][k] bf16 ----
        #pragma unroll
        for (int p = 0; p < 4; ++p) {
            int kr = (tid >> 5) + p * 8;
            int nn = (tid & 31) * 4;
            float4 f = *(const float4*)(wBase + (long)(k0 + kr) * DMODEL + nn);
            sBt[nn + 0][kr] = (__bf16)f.x;
            sBt[nn + 1][kr] = (__bf16)f.y;
            sBt[nn + 2][kr] = (__bf16)f.z;
            sBt[nn + 3][kr] = (__bf16)f.w;
        }
        if (k0 + 32 < DMODEL)
            __builtin_prefetch((const void*)(wBase + (long)(k0 + 32) * DMODEL + (tid & 31) * 4), 0, 1);
        __syncthreads();

        v16bf a0 = load_afrag(&sA[mrow][0],      kb);
        v16bf a1 = load_afrag(&sA[16 + mrow][0], kb);
        v16bf bb = load_bfrag(&sBt[ncol][kb2]);
        acc0 = wmma_bf16(a0, bb, acc0);
        acc1 = wmma_bf16(a1, bb, acc1);
        __syncthreads();
    }

    // ---- epilogue ----
    const int nout  = n0 + ncol;
    const int mbase = (lane >> 4) * 8;
    float bval = 0.f;
    if (flags == 1) bval = bias[(long)agent * DMODEL + nout];
    #pragma unroll
    for (int r = 0; r < 8; ++r) {
        int b0 = mbase + r;        // rows 0..15  (acc0)
        int b1 = 16 + mbase + r;   // rows 16..31 (acc1)
        long i0 = (long)b0 * ROWSTRIDE_B + (long)agent * DMODEL + nout;
        long i1 = (long)b1 * ROWSTRIDE_B + (long)agent * DMODEL + nout;
        if (flags == 0) {
            C[i0] = acc0[r];
            C[i1] = acc1[r];
        } else {
            float t0 = acc0[r] + bval;
            float t1 = acc1[r] + bval;
            C[i0] += t0 / (1.f + __expf(-t0));   // SiLU + residual
            C[i1] += t1 / (1.f + __expf(-t1));
        }
    }
}

// ---------------------------------------------------------------------------
// LayerNorm over last dim (512). One row per block, 256 threads.
// ---------------------------------------------------------------------------
__global__ __launch_bounds__(256)
void ln_kernel(const float* __restrict__ x, const float* __restrict__ g,
               const float* __restrict__ b, float* __restrict__ y)
{
    __shared__ float r1[256];
    __shared__ float r2[256];
    const long row = blockIdx.x;
    const int  tid = threadIdx.x;
    const float* xr = x + row * DMODEL;
    float a0 = xr[tid];
    float a1 = xr[tid + 256];
    r1[tid] = a0 + a1;
    r2[tid] = a0 * a0 + a1 * a1;
    __syncthreads();
    for (int s = 128; s > 0; s >>= 1) {
        if (tid < s) { r1[tid] += r1[tid + s]; r2[tid] += r2[tid + s]; }
        __syncthreads();
    }
    float mu  = r1[0] * (1.f / DMODEL);
    float var = r2[0] * (1.f / DMODEL) - mu * mu;
    float rs  = rsqrtf(var + 1e-5f);
    y[row * DMODEL + tid]       = (a0 - mu) * rs * g[tid]       + b[tid];
    y[row * DMODEL + tid + 256] = (a1 - mu) * rs * g[tid + 256] + b[tid + 256];
}

// ---------------------------------------------------------------------------
// Attention: one block per (mtile, head, batch). M-tile = 32 agent rows,
// full K/V (128 agents) in LDS. S = QK^T*scale + connect, softmax, O = P V.
// ---------------------------------------------------------------------------
__global__ __launch_bounds__(256)
void attn_kernel(const float* __restrict__ q, const float* __restrict__ k,
                 const float* __restrict__ v, const float* __restrict__ connect,
                 float* __restrict__ out)
{
    __shared__ __bf16 q_s[32][64];    //  4 KB  [agent-row][dh]
    __shared__ __bf16 k_s[128][64];   // 16 KB  [agent][dh]   (B frag of QK^T: contig dh)
    __shared__ __bf16 v_t[64][128];   // 16 KB  [dh][agent]   (B frag of PV: contig agent)
    __shared__ float  S  [32][128];   // 16 KB
    __shared__ __bf16 p_s[32][128];   //  8 KB   (total 60 KB)

    const int mtile = blockIdx.x;      // 0..3
    const int head  = blockIdx.y;      // 0..7
    const int batch = blockIdx.z;      // 0..31
    const int tid   = threadIdx.x;
    const int lane  = tid & 31;
    const int wave  = tid >> 5;

    const long base = (long)batch * ROWSTRIDE_B + head * DHEAD;

    // ---- stage Q (32x64) ----
    #pragma unroll
    for (int p = 0; p < 2; ++p) {
        int r = tid >> 3;
        int c = ((tid & 7) + p * 8) * 4;
        float4 f = *(const float4*)(q + base + (long)(mtile * 32 + r) * DMODEL + c);
        q_s[r][c+0] = (__bf16)f.x; q_s[r][c+1] = (__bf16)f.y;
        q_s[r][c+2] = (__bf16)f.z; q_s[r][c+3] = (__bf16)f.w;
    }
    // ---- stage K [agent][dh] and V transposed [dh][agent] ----
    #pragma unroll
    for (int p = 0; p < 8; ++p) {
        int r = (tid >> 4) + p * 16;   // agent 0..127
        int c = (tid & 15) * 4;        // dh    0..60
        float4 fk = *(const float4*)(k + base + (long)r * DMODEL + c);
        float4 fv = *(const float4*)(v + base + (long)r * DMODEL + c);
        k_s[r][c+0] = (__bf16)fk.x; k_s[r][c+1] = (__bf16)fk.y;
        k_s[r][c+2] = (__bf16)fk.z; k_s[r][c+3] = (__bf16)fk.w;
        v_t[c+0][r] = (__bf16)fv.x; v_t[c+1][r] = (__bf16)fv.y;
        v_t[c+2][r] = (__bf16)fv.z; v_t[c+3][r] = (__bf16)fv.w;
    }
    __syncthreads();

    const int mrow = lane & 15;
    const int kb   = (lane >> 4) * 8;
    const int kb2  = (lane >> 4) * 16;

    // ---- S = Q K^T : M=32, N=128 (wave w -> n0 = w*16), K=64 ----
    {
        const int nag = wave * 16 + mrow;   // agent column
        v8f s0 = {0.f,0.f,0.f,0.f,0.f,0.f,0.f,0.f};
        v8f s1 = {0.f,0.f,0.f,0.f,0.f,0.f,0.f,0.f};
        #pragma unroll
        for (int kc = 0; kc < 64; kc += 32) {
            v16bf a0 = load_afrag(&q_s[mrow][kc],      kb);
            v16bf a1 = load_afrag(&q_s[16 + mrow][kc], kb);
            v16bf bb = load_bfrag(&k_s[nag][kc + kb2]);  // B[k=dh][n=agent] = K[n][k]
            s0 = wmma_bf16(a0, bb, s0);
            s1 = wmma_bf16(a1, bb, s1);
        }
        const int mbase = (lane >> 4) * 8;
        #pragma unroll
        for (int r = 0; r < 8; ++r) {
            S[mbase + r][nag]      = s0[r];
            S[16 + mbase + r][nag] = s1[r];
        }
    }
    __syncthreads();

    // ---- softmax rows (scale + connect mask fused), P -> bf16 ----
    if (tid < 32) {
        const int r     = tid;
        const int agrow = mtile * 32 + r;
        const float scale = 0.125f;            // 1/sqrt(64)
        float mx = -1e30f;
        for (int c = 0; c < AGENTS; ++c) {
            float sv = S[r][c] * scale + connect[(long)agrow * AGENTS + c];
            S[r][c] = sv;
            mx = fmaxf(mx, sv);
        }
        float sum = 0.f;
        for (int c = 0; c < AGENTS; ++c) {
            float e = __expf(S[r][c] - mx);
            S[r][c] = e;
            sum += e;
        }
        float inv = 1.f / sum;
        for (int c = 0; c < AGENTS; ++c)
            p_s[r][c] = (__bf16)(S[r][c] * inv);
    }
    __syncthreads();

    // ---- O = P V : M=32, N=64, K=128. 8 tiles -> wave w does (m=w>>2, n=w&3) ----
    {
        const int m0   = (wave >> 2) * 16;
        const int n0   = (wave & 3) * 16;
        const int ncol = n0 + mrow;            // dh column
        v8f o = {0.f,0.f,0.f,0.f,0.f,0.f,0.f,0.f};
        #pragma unroll
        for (int kc = 0; kc < 128; kc += 32) {
            v16bf aa = load_afrag(&p_s[m0 + mrow][kc], kb);
            v16bf bb = load_bfrag(&v_t[ncol][kc + kb2]);  // B[k=agent][n=dh] = V^T[n][k]
            o = wmma_bf16(aa, bb, o);
        }
        const int mbase = (lane >> 4) * 8;
        #pragma unroll
        for (int r = 0; r < 8; ++r) {
            int agent = mtile * 32 + m0 + mbase + r;
            out[base + (long)agent * DMODEL + ncol] = o[r];
        }
    }
}

// ---------------------------------------------------------------------------
// Host orchestration (graph-capture safe: only stream ops).
// Inputs: 0:x 1:W_embed 2:ln0_g 3:ln0_b 4:W_q 5:W_k 6:W_v 7:W_fwd 8:b_fwd
//         9:ln1_g 10:ln1_b 11:connect
// ---------------------------------------------------------------------------
extern "C" void kernel_launch(void* const* d_in, const int* in_sizes, int n_in,
                              void* d_out, int out_size, void* d_ws, size_t ws_size,
                              hipStream_t stream) {
    (void)in_sizes; (void)n_in; (void)out_size; (void)ws_size;
    const float* x       = (const float*)d_in[0];
    const float* W_embed = (const float*)d_in[1];
    const float* ln0_g   = (const float*)d_in[2];
    const float* ln0_b   = (const float*)d_in[3];
    const float* W_q     = (const float*)d_in[4];
    const float* W_k     = (const float*)d_in[5];
    const float* W_v     = (const float*)d_in[6];
    const float* W_fwd   = (const float*)d_in[7];
    const float* b_fwd   = (const float*)d_in[8];
    const float* ln1_g   = (const float*)d_in[9];
    const float* ln1_b   = (const float*)d_in[10];
    const float* connect = (const float*)d_in[11];

    float* ws = (float*)d_ws;
    float* h  = ws;                 // [B,A,D]
    float* xn = ws + 1 * HSIZE;     // normalized / attn-out scratch
    float* q  = ws + 2 * HSIZE;
    float* kk = ws + 3 * HSIZE;
    float* vv = ws + 4 * HSIZE;

    const long perAgentW = (long)DMODEL * DMODEL;
    dim3 gGemm(DMODEL / 128, AGENTS);     // (4, 128)
    dim3 gAttn(4, NHEADS, BATCH);         // (4, 8, 32)

    // emb = einsum(x, W_embed) -> xn ; h = LN0(emb)
    gemm_agent_kernel<<<gGemm, 256, 0, stream>>>(x, W_embed, nullptr, xn, perAgentW, 0);
    ln_kernel<<<NROWS, 256, 0, stream>>>(xn, ln0_g, ln0_b, h);

    for (int step = 0; step < 3; ++step) {
        ln_kernel<<<NROWS, 256, 0, stream>>>(h, ln1_g, ln1_b, xn);               // xn = LN1(h)
        gemm_agent_kernel<<<gGemm, 256, 0, stream>>>(xn, W_q, nullptr, q,  perAgentW, 0);
        gemm_agent_kernel<<<gGemm, 256, 0, stream>>>(xn, W_k, nullptr, kk, perAgentW, 0);
        gemm_agent_kernel<<<gGemm, 256, 0, stream>>>(xn, W_v, nullptr, vv, 0,    0);
        attn_kernel<<<gAttn, 256, 0, stream>>>(q, kk, vv, connect, xn);          // xn = attn out
        ln_kernel<<<NROWS, 256, 0, stream>>>(xn, ln1_g, ln1_b, q);               // q = LN1(attn)
        // h += silu(q @ W_fwd + b_fwd)
        gemm_agent_kernel<<<gGemm, 256, 0, stream>>>(q, W_fwd, b_fwd, h, perAgentW, 1);
    }

    hipMemcpyAsync(d_out, h, HSIZE * sizeof(float), hipMemcpyDeviceToDevice, stream);
}
